// TargetPatchEncoder_81200651698710
// MI455X (gfx1250) — matrix operand order repaired and checked by reading, not compiled, not run on hardware
//
#include <hip/hip_runtime.h>
#include <hip/hip_bf16.h>

// ---------------------------------------------------------------------------
// MI455X (gfx1250) implementation.
//  * All large GEMMs run on v_wmma_f32_16x16x32_bf16 (bf16 in, fp32 accum).
//  * Tiles stream into LDS via GLOBAL_LOAD_ASYNC_TO_LDS_B128 (ASYNCcnt),
//    overlapped with WMMA issue; 2-stage LDS double buffer.
//  * Residual stream kept fp32; GEMM operands converted to bf16 on the fly.
//  * Workspace requirement: ~470 MB (lifetime-aliased regions, see offsets).
// ---------------------------------------------------------------------------

typedef __attribute__((ext_vector_type(16))) __bf16 v16bf;
typedef __attribute__((ext_vector_type(8)))  float  v8f;

#define D_MODEL 1024
#define TOKENS  32768      // B*NP*PS = 4*512*16
#define PATCHES 2048       // B*NP
#define PS      16
#define NH      4
#define HD      256
#define PH      4
#define PHD     256

#define RMS_EPS 1.1920929e-7f

__device__ __forceinline__ unsigned short f32_to_bf16(float f) {
  union { float f; unsigned u; } v; v.f = f;
  unsigned u = v.u;
  u += 0x7FFFu + ((u >> 16) & 1u);   // round to nearest even
  return (unsigned short)(u >> 16);
}
__device__ __forceinline__ float bf16_to_f32(unsigned short h) {
  union { unsigned u; float f; } v; v.u = ((unsigned)h) << 16; return v.f;
}
__device__ __forceinline__ float siluf(float x) { return x / (1.f + __expf(-x)); }

// LDS byte address from a flat pointer to __shared__ memory.
// ISA 10.2: the LDS address is the low 32 bits of the flat address.
__device__ __forceinline__ unsigned lds_off(const void* p) {
  return (unsigned)(unsigned long long)p;
}

// ---------------------------------------------------------------------------
// Generic bf16 WMMA GEMM:  C[M,N] = A[M,K] @ W[N,K]^T   (fp32 accumulate)
// Block tile 128x128, K-step 32, 8 waves (wave tile 64x32 = 4x2 WMMA tiles).
// LDS double buffered; tiles fetched with async global->LDS DMA (ASYNCcnt),
// fully overlapped with the 8 WMMAs of the live tile.
// Epilogue MODE: 0 f32 store | 1 bf16 store | 2 silu->bf16
//               3 silu(AUX)*c -> f32 | 4 AUX + c -> f32
// ---------------------------------------------------------------------------
#define BM 128
#define BN 128
#define BK 32
#define LDT 40   // padded LDS row stride in halves (80B, 16B-aligned)

union FragBf { v16bf v; uint4 q[2]; unsigned short h[16]; };

template <int MODE>
__global__ __launch_bounds__(256) void gemm_bf16_wmma(
    const unsigned short* __restrict__ A,   // M x K, bf16 row-major
    const unsigned short* __restrict__ W,   // N x K, bf16 row-major
    const float* __restrict__ AUX,          // optional fp32 M x N
    void* __restrict__ OUT,
    int M, int N, int K)
{
  __shared__ unsigned short sA[2][BM * LDT];
  __shared__ unsigned short sW[2][BN * LDT];

  const int tid  = threadIdx.x;
  const int bm   = blockIdx.y * BM;
  const int bn   = blockIdx.x * BN;
  const int lane = tid & 31;
  const int wave = tid >> 5;
  const int wm   = (wave & 1) * 64;       // wave M offset
  const int wn   = (wave >> 1) * 32;      // wave N offset

  // global->LDS mapping: 512 16B chunks per tile, 2 per thread (coalesced)
  const int r0 = tid >> 2;                 // rows 0..63
  const int c0 = (tid & 3) * 8;            // half-column within row
  const int r1 = r0 + 64;                  // rows 64..127

  v8f acc[4][2];
  #pragma unroll
  for (int i = 0; i < 4; ++i)
    #pragma unroll
    for (int j = 0; j < 2; ++j)
      #pragma unroll
      for (int r = 0; r < 8; ++r) acc[i][j][r] = 0.f;

  const int NK = K / BK;
  const unsigned long long baseA = (unsigned long long)A;
  const unsigned long long baseW = (unsigned long long)W;

  // Async DMA of one 128x32 A tile + 128x32 W tile into LDS stage `buf`.
  // GVS mode: per-lane signed 32-bit byte offset from SGPR64 base (<64MB here).
  auto aload = [&](int kt, int buf) {
    const int ga0 = ((bm + r0) * K + kt * BK + c0) * 2;
    const int ga1 = ((bm + r1) * K + kt * BK + c0) * 2;
    const int gw0 = ((bn + r0) * K + kt * BK + c0) * 2;
    const int gw1 = ((bn + r1) * K + kt * BK + c0) * 2;
    const unsigned la0 = lds_off(&sA[buf][r0 * LDT + c0]);
    const unsigned la1 = lds_off(&sA[buf][r1 * LDT + c0]);
    const unsigned lw0 = lds_off(&sW[buf][r0 * LDT + c0]);
    const unsigned lw1 = lds_off(&sW[buf][r1 * LDT + c0]);
    asm volatile("global_load_async_to_lds_b128 %0, %1, %2"
                 :: "v"(la0), "v"(ga0), "s"(baseA) : "memory");
    asm volatile("global_load_async_to_lds_b128 %0, %1, %2"
                 :: "v"(la1), "v"(ga1), "s"(baseA) : "memory");
    asm volatile("global_load_async_to_lds_b128 %0, %1, %2"
                 :: "v"(lw0), "v"(gw0), "s"(baseW) : "memory");
    asm volatile("global_load_async_to_lds_b128 %0, %1, %2"
                 :: "v"(lw1), "v"(gw1), "s"(baseW) : "memory");
  };

  aload(0, 0);
  asm volatile("s_wait_asynccnt 0x0" ::: "memory");
  __syncthreads();

  const int m16 = lane & 15;
  const int kbA = (lane < 16) ? 0 : 8;    // ISA A-layout K base per lane half
  const int kbB = (lane < 16) ? 0 : 8;    // ISA B-layout K base per lane half

  for (int kt = 0; kt < NK; ++kt) {
    const int cur = kt & 1;
    if (kt + 1 < NK) aload(kt + 1, cur ^ 1);         // DMA next tile (async)
    if (kt + 2 < NK) {                               // L2 prefetch hint
      __builtin_prefetch(A + (size_t)(bm + r0) * K + (size_t)(kt + 2) * BK + c0, 0, 1);
      __builtin_prefetch(W + (size_t)(bn + r0) * K + (size_t)(kt + 2) * BK + c0, 0, 1);
    }

    FragBf af[4], bf2[2];
    #pragma unroll
    for (int i = 0; i < 4; ++i) {
      const int m = wm + i * 16 + m16;
      af[i].q[0] = *(const uint4*)&sA[cur][m * LDT + kbA];       // K = kb..kb+7
      af[i].q[1] = *(const uint4*)&sA[cur][m * LDT + kbA + 16];  // K = kb+16..kb+23
    }
    #pragma unroll
    for (int j = 0; j < 2; ++j) {
      const int n = wn + j * 16 + m16;
      bf2[j].q[0] = *(const uint4*)&sW[cur][n * LDT + kbB];      // K = kb..kb+7
      bf2[j].q[1] = *(const uint4*)&sW[cur][n * LDT + kbB + 16]; // K = kb+16..kb+23
    }
    #pragma unroll
    for (int i = 0; i < 4; ++i)
      #pragma unroll
      for (int j = 0; j < 2; ++j)
        acc[i][j] = __builtin_amdgcn_wmma_f32_16x16x32_bf16(
            false, af[i].v, false, bf2[j].v, (short)0, acc[i][j], false, false);

    if (kt + 1 < NK) asm volatile("s_wait_asynccnt 0x0" ::: "memory");
    __syncthreads();
  }

  // Epilogue: C/D layout — lane = column group, VGPR r -> row (lane<16 ? r : r+8)
  const int n16 = lane & 15;
  const int mhi = (lane < 16) ? 0 : 8;
  #pragma unroll
  for (int i = 0; i < 4; ++i) {
    #pragma unroll
    for (int j = 0; j < 2; ++j) {
      #pragma unroll
      for (int r = 0; r < 8; ++r) {
        const int gm = bm + wm + i * 16 + mhi + r;
        const int gn = bn + wn + j * 16 + n16;
        const size_t idx = (size_t)gm * N + gn;
        const float c = acc[i][j][r];
        if      (MODE == 0) ((float*)OUT)[idx] = c;
        else if (MODE == 1) ((unsigned short*)OUT)[idx] = f32_to_bf16(c);
        else if (MODE == 2) ((unsigned short*)OUT)[idx] = f32_to_bf16(siluf(c));
        else if (MODE == 3) ((float*)OUT)[idx] = siluf(AUX[idx]) * c;
        else                ((float*)OUT)[idx] = AUX[idx] + c;
      }
    }
  }
}

// ---------------------------------------------------------------------------
// Elementwise / normalization / conv / attention / pooling kernels
// ---------------------------------------------------------------------------
__global__ __launch_bounds__(256) void k_f2bf(const float* __restrict__ in,
                                              unsigned short* __restrict__ out, int n) {
  int i = blockIdx.x * 256 + threadIdx.x;
  if (i < n) out[i] = f32_to_bf16(in[i]);
}

// x = rms_norm(patch + local_pos[p])  -> bf16
__global__ __launch_bounds__(256) void k_prep(const float* __restrict__ patch,
                                              const float* __restrict__ lpos,
                                              unsigned short* __restrict__ out) {
  __shared__ float red[256];
  const size_t row = blockIdx.x;
  const int p = (int)(row & (PS - 1));
  const float* in = patch + row * D_MODEL;
  const float* lp = lpos + (size_t)p * D_MODEL;
  const int tid = threadIdx.x;
  float v[4]; float ss = 0.f;
  #pragma unroll
  for (int i = 0; i < 4; ++i) { int d = tid + i * 256; v[i] = in[d] + lp[d]; ss += v[i] * v[i]; }
  red[tid] = ss; __syncthreads();
  for (int s = 128; s > 0; s >>= 1) { if (tid < s) red[tid] += red[tid + s]; __syncthreads(); }
  const float sc = rsqrtf(red[0] * (1.f / D_MODEL) + RMS_EPS);
  #pragma unroll
  for (int i = 0; i < 4; ++i)
    out[row * D_MODEL + tid + i * 256] = f32_to_bf16(v[i] * sc);
}

// rms_norm over last dim (1024); BF16OUT selects output type
template <bool BF16OUT>
__global__ __launch_bounds__(256) void k_rms(const float* __restrict__ in_,
                                             void* __restrict__ out_) {
  __shared__ float red[256];
  const size_t row = blockIdx.x;
  const float* in = in_ + row * D_MODEL;
  const int tid = threadIdx.x;
  float v[4]; float ss = 0.f;
  #pragma unroll
  for (int i = 0; i < 4; ++i) { v[i] = in[tid + i * 256]; ss += v[i] * v[i]; }
  red[tid] = ss; __syncthreads();
  for (int s = 128; s > 0; s >>= 1) { if (tid < s) red[tid] += red[tid + s]; __syncthreads(); }
  const float sc = rsqrtf(red[0] * (1.f / D_MODEL) + RMS_EPS);
  #pragma unroll
  for (int i = 0; i < 4; ++i) {
    const float o = v[i] * sc;
    if (BF16OUT) ((unsigned short*)out_)[row * D_MODEL + tid + i * 256] = f32_to_bf16(o);
    else         ((float*)out_)[row * D_MODEL + tid + i * 256] = o;
  }
}

// depthwise causal conv (K=5, left pad 4) fused with residual: y = x + conv(x)
__global__ __launch_bounds__(256) void k_conv(const float* __restrict__ X,
                                              const float* __restrict__ CW,
                                              float* __restrict__ Y) {
  const int g = blockIdx.x * 256 + threadIdx.x;        // (bn, d) column
  const int bn = g >> 10;
  const int d  = g & (D_MODEL - 1);
  float w[5];
  #pragma unroll
  for (int j = 0; j < 5; ++j) w[j] = CW[d * 5 + j];
  const float* x = X + (size_t)bn * PS * D_MODEL + d;
  float xv[PS];
  #pragma unroll
  for (int p = 0; p < PS; ++p) xv[p] = x[p * D_MODEL];
  float* y = Y + (size_t)bn * PS * D_MODEL + d;
  #pragma unroll
  for (int p = 0; p < PS; ++p) {
    float acc = xv[p];
    #pragma unroll
    for (int j = 0; j < 5; ++j) {
      const int src = p - 4 + j;
      if (src >= 0) acc += w[j] * xv[src];
    }
    y[p * D_MODEL] = acc;
  }
}

// per (patch, head) 16x16 softmax attention; qkv bf16 in, sa bf16 out (~1% FLOPs)
__global__ __launch_bounds__(256) void k_attn(const unsigned short* __restrict__ QKV,
                                              unsigned short* __restrict__ SA) {
  __shared__ float sq[PS][HD + 1], sk[PS][HD + 1], sv[PS][HD + 1];
  __shared__ float sS[PS][17], sP[PS][16];
  const int bh = blockIdx.x;
  const int bn = bh >> 2;
  const int h  = bh & 3;
  const unsigned short* base = QKV + (size_t)bn * PS * 3 * D_MODEL + h * HD;
  const int tid = threadIdx.x;
  for (int idx = tid; idx < PS * HD; idx += 256) {
    const int p = idx >> 8, d = idx & (HD - 1);
    const unsigned short* row = base + (size_t)p * 3 * D_MODEL + d;
    sq[p][d] = bf16_to_f32(row[0]);
    sk[p][d] = bf16_to_f32(row[D_MODEL]);
    sv[p][d] = bf16_to_f32(row[2 * D_MODEL]);
  }
  __syncthreads();
  {
    const int qi = tid >> 4, ki = tid & 15;
    float acc = 0.f;
    for (int d = 0; d < HD; ++d) acc += sq[qi][d] * sk[ki][d];
    sS[qi][ki] = acc * 0.0625f;               // HD^-0.5
  }
  __syncthreads();
  if (tid < PS) {
    float mx = -3.4e38f;
    for (int k = 0; k < 16; ++k) mx = fmaxf(mx, sS[tid][k]);
    float e[16], s = 0.f;
    for (int k = 0; k < 16; ++k) { e[k] = __expf(sS[tid][k] - mx); s += e[k]; }
    const float inv = 1.f / s;
    for (int k = 0; k < 16; ++k) sP[tid][k] = e[k] * inv;
  }
  __syncthreads();
  unsigned short* out = SA + (size_t)bn * PS * D_MODEL + h * HD;
  for (int idx = tid; idx < PS * HD; idx += 256) {
    const int p = idx >> 8, d = idx & (HD - 1);
    float acc = 0.f;
    #pragma unroll
    for (int k = 0; k < 16; ++k) acc += sP[p][k] * sv[k][d];
    out[(size_t)p * D_MODEL + d] = f32_to_bf16(acc);
  }
}

// fused: x_n = rms_norm(x); logits = x_n @ w_pool^T / clip(temp); softmax over p;
// pooled[d] = sum_p w[h(d)][p] * x_n[p][d]
__global__ __launch_bounds__(256) void k_pool(const float* __restrict__ X,
                                              const float* __restrict__ WPOOL,
                                              const float* __restrict__ TEMP,
                                              float* __restrict__ POOLED) {
  __shared__ float red[PS][17];
  __shared__ float scale[PS];
  __shared__ float wgt[PH][PS];
  const int bn = blockIdx.x;
  const float* x = X + (size_t)bn * PS * D_MODEL;
  const int tid = threadIdx.x;
  const int p = tid >> 4, l = tid & 15;
  float ss = 0.f;
  for (int d = l; d < D_MODEL; d += 16) { const float v = x[p * D_MODEL + d]; ss += v * v; }
  red[p][l] = ss;
  __syncthreads();
  if (tid < PS) {
    float s = 0.f;
    for (int i = 0; i < 16; ++i) s += red[tid][i];
    scale[tid] = rsqrtf(s * (1.f / D_MODEL) + RMS_EPS);
  }
  __syncthreads();
  if (tid < PS * PH) {               // logits
    const int pp = tid >> 2, hp = tid & 3;
    float acc = 0.f;
    for (int d = 0; d < D_MODEL; ++d) acc += x[pp * D_MODEL + d] * WPOOL[hp * D_MODEL + d];
    float t = TEMP[0]; t = (t < 0.01f) ? 0.01f : t;
    red[hp][pp] = acc * scale[pp] / t;
  }
  __syncthreads();
  if (tid < PH) {                    // softmax over positions
    float mx = -3.4e38f;
    for (int i = 0; i < 16; ++i) mx = fmaxf(mx, red[tid][i]);
    float e[16], s = 0.f;
    for (int i = 0; i < 16; ++i) { e[i] = __expf(red[tid][i] - mx); s += e[i]; }
    const float inv = 1.f / s;
    for (int i = 0; i < 16; ++i) wgt[tid][i] = e[i] * inv;
  }
  __syncthreads();
  for (int d = tid; d < D_MODEL; d += 256) {
    const int hp = d >> 8;           // PHD = 256
    float acc = 0.f;
    #pragma unroll
    for (int pp = 0; pp < 16; ++pp) acc += wgt[hp][pp] * (x[pp * D_MODEL + d] * scale[pp]);
    POOLED[(size_t)bn * D_MODEL + d] = acc;
  }
}

// ---------------------------------------------------------------------------
// Host launch
// ---------------------------------------------------------------------------
extern "C" void kernel_launch(void* const* d_in, const int* in_sizes, int n_in,
                              void* d_out, int out_size, void* d_ws, size_t ws_size,
                              hipStream_t stream) {
  const float* patch  = (const float*)d_in[0];
  const float* lpos   = (const float*)d_in[1];
  const float* w_gate = (const float*)d_in[2];
  const float* w_in   = (const float*)d_in[3];
  const float* conv_w = (const float*)d_in[4];
  const float* w_qkv  = (const float*)d_in[5];
  const float* w_io   = (const float*)d_in[6];
  const float* w_pool = (const float*)d_in[7];
  const float* temp   = (const float*)d_in[8];
  const float* w_up   = (const float*)d_in[9];
  const float* w_down = (const float*)d_in[10];
  const float* w_out  = (const float*)d_in[11];
  float* out = (float*)d_out;

  char* ws = (char*)d_ws;
  size_t off = 0;
  auto nxt = [&](size_t b) { char* p = ws + off; off += (b + 255) & ~(size_t)255; return p; };

  const size_t DD = (size_t)D_MODEL * D_MODEL;
  unsigned short* wb_gate = (unsigned short*)nxt(DD * 2);
  unsigned short* wb_in   = (unsigned short*)nxt(DD * 2);
  unsigned short* wb_qkv  = (unsigned short*)nxt(3 * DD * 2);
  unsigned short* wb_io   = (unsigned short*)nxt(DD * 2);
  unsigned short* wb_up   = (unsigned short*)nxt(DD * 2);
  unsigned short* wb_down = (unsigned short*)nxt(DD * 2);
  unsigned short* wb_out  = (unsigned short*)nxt(DD * 2);

  // xb: normalized activations (bf16); later reused for x_sa, then for sa output
  unsigned short* xb = (unsigned short*)nxt((size_t)TOKENS * D_MODEL * 2);
  unsigned short* sab = xb;                               // alias (lifetimes disjoint)
  // R2 (256 MiB): gate fp32 | x_mlp fp32  ->  qkv bf16  ->  x_final fp32 + small bufs
  char* R2 = nxt((size_t)256 << 20);
  float* g_aux  = (float*)R2;                             // gate pre-activation
  float* x_mlp  = (float*)(R2 + ((size_t)128 << 20));     // MLP output
  unsigned short* qkvb = (unsigned short*)R2;             // 192 MiB (gate/x_mlp dead)
  float* x_fin  = (float*)R2;                             // post-attention residual
  char* small   = R2 + ((size_t)TOKENS * D_MODEL * 4);    // beyond x_fin (134 MiB)
  float* pooled = (float*)(small);
  unsigned short* pb = (unsigned short*)(small + ((size_t)PATCHES * D_MODEL * 4));
  unsigned short* ub = (unsigned short*)(small + ((size_t)PATCHES * D_MODEL * 6));
  float* yf = (float*)(small + ((size_t)PATCHES * D_MODEL * 8));
  unsigned short* yb = (unsigned short*)(small + ((size_t)PATCHES * D_MODEL * 12));
  float* zf = (float*)(small + ((size_t)PATCHES * D_MODEL * 14));
  // bufB: x after conv residual (fp32)
  float* x_conv = (float*)nxt((size_t)TOKENS * D_MODEL * 4);

  const int nDD = (int)DD;
  const dim3 blk(256);
  // --- weights -> bf16 ---
  k_f2bf<<<(nDD + 255) / 256, blk, 0, stream>>>(w_gate, wb_gate, nDD);
  k_f2bf<<<(nDD + 255) / 256, blk, 0, stream>>>(w_in,   wb_in,   nDD);
  k_f2bf<<<(3 * nDD + 255) / 256, blk, 0, stream>>>(w_qkv, wb_qkv, 3 * nDD);
  k_f2bf<<<(nDD + 255) / 256, blk, 0, stream>>>(w_io,   wb_io,   nDD);
  k_f2bf<<<(nDD + 255) / 256, blk, 0, stream>>>(w_up,   wb_up,   nDD);
  k_f2bf<<<(nDD + 255) / 256, blk, 0, stream>>>(w_down, wb_down, nDD);
  k_f2bf<<<(nDD + 255) / 256, blk, 0, stream>>>(w_out,  wb_out,  nDD);

  // --- pipeline ---
  k_prep<<<TOKENS, blk, 0, stream>>>(patch, lpos, xb);
  gemm_bf16_wmma<0><<<dim3(8, 256), blk, 0, stream>>>(xb, wb_gate, nullptr, g_aux,
                                                      TOKENS, D_MODEL, D_MODEL);
  gemm_bf16_wmma<3><<<dim3(8, 256), blk, 0, stream>>>(xb, wb_in, g_aux, x_mlp,
                                                      TOKENS, D_MODEL, D_MODEL);
  k_conv<<<(TOKENS / PS) * D_MODEL / 256, blk, 0, stream>>>(x_mlp, conv_w, x_conv);
  k_rms<true><<<TOKENS, blk, 0, stream>>>(x_conv, xb);            // x_sa (bf16)
  gemm_bf16_wmma<1><<<dim3(24, 256), blk, 0, stream>>>(xb, wb_qkv, nullptr, qkvb,
                                                       TOKENS, 3 * D_MODEL, D_MODEL);
  k_attn<<<PATCHES * NH, blk, 0, stream>>>(qkvb, sab);
  gemm_bf16_wmma<4><<<dim3(8, 256), blk, 0, stream>>>(sab, wb_io, x_conv, x_fin,
                                                      TOKENS, D_MODEL, D_MODEL);
  k_pool<<<PATCHES, blk, 0, stream>>>(x_fin, w_pool, temp, pooled);
  k_rms<true><<<PATCHES, blk, 0, stream>>>(pooled, pb);
  gemm_bf16_wmma<2><<<dim3(8, 16), blk, 0, stream>>>(pb, wb_up, nullptr, ub,
                                                     PATCHES, D_MODEL, D_MODEL);
  gemm_bf16_wmma<4><<<dim3(8, 16), blk, 0, stream>>>(ub, wb_down, pooled, yf,
                                                     PATCHES, D_MODEL, D_MODEL);
  k_f2bf<<<(PATCHES * D_MODEL + 255) / 256, blk, 0, stream>>>(yf, yb, PATCHES * D_MODEL);
  gemm_bf16_wmma<0><<<dim3(8, 16), blk, 0, stream>>>(yb, wb_out, nullptr, zf,
                                                     PATCHES, D_MODEL, D_MODEL);
  k_rms<false><<<2048, blk, 0, stream>>>(zf, out);
}
